// GatedGraphNeuralNetwork_9526237462875
// MI455X (gfx1250) — matrix operand order repaired and checked by reading, not compiled, not run on hardware
//
#include <hip/hip_runtime.h>

#define HD 512
#define TD 4
#define ED 50000
#define ND 20000

typedef __attribute__((ext_vector_type(16))) __bf16 v16bf;
typedef __attribute__((ext_vector_type(8)))  __bf16 v8bf;
typedef __attribute__((ext_vector_type(8)))  float  v8f;

#define WMMA_BF16(A, B, C) \
  __builtin_amdgcn_wmma_f32_16x16x32_bf16(false, (A), false, (B), (short)0, (C), false, false)

// A-fragment (16x32 bf16): lane holds contiguous K = {0..15} (lanes 0-15) or {16..31}
// caller passes p already offset by kk + 16*(lane>>4); chunks at +0 and +8 elements.
__device__ __forceinline__ v16bf ldfragA(const unsigned short* p) {
  v8bf c0 = *(const v8bf*)p;
  v8bf c1 = *(const v8bf*)(p + 8);
  return __builtin_shufflevector(c0, c1, 0,1,2,3,4,5,6,7,8,9,10,11,12,13,14,15);
}
// B-fragment (32x16 bf16): lane holds contiguous K = {0..15} or {16..31}
// caller passes p already offset by kk + 16*(lane>>4); chunks at +0 and +8 elements.
__device__ __forceinline__ v16bf ldfragB(const unsigned short* p) {
  v8bf c0 = *(const v8bf*)p;
  v8bf c1 = *(const v8bf*)(p + 8);
  return __builtin_shufflevector(c0, c1, 0,1,2,3,4,5,6,7,8,9,10,11,12,13,14,15);
}

__device__ __forceinline__ void atomAddF(float* p, float v) {
  __hip_atomic_fetch_add(p, v, __ATOMIC_RELAXED, __HIP_MEMORY_SCOPE_AGENT);
}

// fp32 -> bf16 hi/lo split (RNE both halves)
__global__ __launch_bounds__(256) void k_cvt_split(const float* __restrict__ src,
                                                   unsigned short* __restrict__ hi,
                                                   unsigned short* __restrict__ lo, long n) {
  long i = (long)blockIdx.x * blockDim.x + threadIdx.x;
  if (i >= n) return;
  float f = src[i];
  unsigned u = __float_as_uint(f);
  unsigned r = u + 0x7FFFu + ((u >> 16) & 1u);
  unsigned short hb = (unsigned short)(r >> 16);
  float l = f - __uint_as_float((unsigned)hb << 16);
  unsigned ul = __float_as_uint(l);
  unsigned rl = ul + 0x7FFFu + ((ul >> 16) & 1u);
  hi[i] = hb;
  lo[i] = (unsigned short)(rl >> 16);
}

__global__ __launch_bounds__(256) void k_fill0(float* __restrict__ p, long n) {
  long i = (long)blockIdx.x * blockDim.x + threadIdx.x;
  if (i < n) p[i] = 0.0f;
}

// msgs[t,e,g] = sum_h hbf[src[t,e], h] * W[t,g,h] + b[t,g]; scatter-add into incoming[tgt[t,e], g]
__global__ __launch_bounds__(256) void k_msg(const unsigned short* __restrict__ hhi,
                                             const unsigned short* __restrict__ hlo,
                                             const unsigned short* __restrict__ whi,  // [T][H][H]
                                             const unsigned short* __restrict__ wlo,
                                             const float* __restrict__ bias,          // [T][H]
                                             const int* __restrict__ adj,             // [T][E][2]
                                             float* __restrict__ incoming) {          // [N][H]
  const int lane = threadIdx.x & 31;
  const int wave = threadIdx.x >> 5;
  const int wr = wave & 3, wc = wave >> 1 >> 1;  // wc = wave>>2
  const int t = blockIdx.z;
  const int erow = blockIdx.x * 64 + wr * 16;
  const int colbase = blockIdx.y * 64 + wc * 32;
  const int half = lane >> 4, l16 = lane & 15;
  const int kaoff = half * 16, kboff = half * 16;

  int e = erow + l16; if (e >= ED) e = ED - 1;
  const long aoff = (long)adj[(((long)t * ED) + e) * 2] * HD;
  const unsigned short* arh = hhi + aoff;
  const unsigned short* arl = hlo + aoff;

  const unsigned short* wth = whi + (long)t * HD * HD;
  const unsigned short* wtl = wlo + (long)t * HD * HD;
  const unsigned short* b0h = wth + (long)(colbase + l16) * HD;
  const unsigned short* b0l = wtl + (long)(colbase + l16) * HD;
  const unsigned short* b1h = wth + (long)(colbase + 16 + l16) * HD;
  const unsigned short* b1l = wtl + (long)(colbase + 16 + l16) * HD;

  v8f c0 = {}; v8f c1 = {};
  for (int kk = 0; kk < HD; kk += 32) {
    v16bf Ah = ldfragA(arh + kk + kaoff);
    v16bf Al = ldfragA(arl + kk + kaoff);
    v16bf B0h = ldfragB(b0h + kk + kboff);
    v16bf B0l = ldfragB(b0l + kk + kboff);
    v16bf B1h = ldfragB(b1h + kk + kboff);
    v16bf B1l = ldfragB(b1l + kk + kboff);
    c0 = WMMA_BF16(Ah, B0h, c0);
    c0 = WMMA_BF16(Ah, B0l, c0);
    c0 = WMMA_BF16(Al, B0h, c0);
    c1 = WMMA_BF16(Ah, B1h, c1);
    c1 = WMMA_BF16(Ah, B1l, c1);
    c1 = WMMA_BF16(Al, B1h, c1);
  }

  const float bv0 = bias[(long)t * HD + colbase + l16];
  const float bv1 = bias[(long)t * HD + colbase + 16 + l16];
  int tgts[8];
#pragma unroll
  for (int r = 0; r < 8; ++r) {
    int em = erow + half * 8 + r;
    tgts[r] = (em < ED) ? adj[(((long)t * ED) + em) * 2 + 1] : -1;
  }
#pragma unroll
  for (int r = 0; r < 8; ++r) {
    if (tgts[r] >= 0) {
      float* p = incoming + (long)tgts[r] * HD + colbase;
      atomAddF(p + l16,      c0[r] + bv0);
      atomAddF(p + 16 + l16, c1[r] + bv1);
    }
  }
}

// out[m, n] = sum_k info[m, k] * W[n, k] + bias[n]; info = [A0 (k<ksplit) | A1], A row stride = HD
__global__ __launch_bounds__(256) void k_gemm(const unsigned short* __restrict__ a0hi,
                                              const unsigned short* __restrict__ a0lo,
                                              const unsigned short* __restrict__ a1hi,
                                              const unsigned short* __restrict__ a1lo,
                                              int ksplit, int ktot,
                                              const unsigned short* __restrict__ whi,  // [outC][ktot]
                                              const unsigned short* __restrict__ wlo,
                                              const float* __restrict__ bias,
                                              float* __restrict__ out, int outC, int M) {
  const int lane = threadIdx.x & 31;
  const int wave = threadIdx.x >> 5;
  const int wr = wave & 3, wc = wave >> 2;
  const int mrow = blockIdx.x * 64 + wr * 16;
  const int colbase = blockIdx.y * 64 + wc * 32;
  const int half = lane >> 4, l16 = lane & 15;
  const int kaoff = half * 16, kboff = half * 16;

  int m = mrow + l16; if (m >= M) m = M - 1;
  const long arow = (long)m * HD;
  const unsigned short* a0h = a0hi + arow; const unsigned short* a0l = a0lo + arow;
  const unsigned short* a1h = a1hi + arow; const unsigned short* a1l = a1lo + arow;
  const unsigned short* b0h = whi + (long)(colbase + l16) * ktot;
  const unsigned short* b0l = wlo + (long)(colbase + l16) * ktot;
  const unsigned short* b1h = whi + (long)(colbase + 16 + l16) * ktot;
  const unsigned short* b1l = wlo + (long)(colbase + 16 + l16) * ktot;

  v8f c0 = {}; v8f c1 = {};
  for (int kk = 0; kk < ktot; kk += 32) {
    const unsigned short *ah, *al; int ka;
    if (kk < ksplit) { ah = a0h; al = a0l; ka = kk; }
    else             { ah = a1h; al = a1l; ka = kk - ksplit; }
    v16bf Ah = ldfragA(ah + ka + kaoff);
    v16bf Al = ldfragA(al + ka + kaoff);
    v16bf B0h = ldfragB(b0h + kk + kboff);
    v16bf B0l = ldfragB(b0l + kk + kboff);
    v16bf B1h = ldfragB(b1h + kk + kboff);
    v16bf B1l = ldfragB(b1l + kk + kboff);
    c0 = WMMA_BF16(Ah, B0h, c0);
    c0 = WMMA_BF16(Ah, B0l, c0);
    c0 = WMMA_BF16(Al, B0h, c0);
    c1 = WMMA_BF16(Ah, B1h, c1);
    c1 = WMMA_BF16(Ah, B1l, c1);
    c1 = WMMA_BF16(Al, B1h, c1);
  }

  const float bv0 = bias[colbase + l16];
  const float bv1 = bias[colbase + 16 + l16];
#pragma unroll
  for (int r = 0; r < 8; ++r) {
    int mg = mrow + half * 8 + r;
    if (mg < M) {
      float* p = out + (long)mg * outC + colbase;
      p[l16]      = c0[r] + bv0;
      p[16 + l16] = c1[r] + bv1;
    }
  }
}

__global__ __launch_bounds__(256) void k_gru(const float* __restrict__ gi,
                                             const float* __restrict__ gh,
                                             float* __restrict__ h, long n) {
  long i = (long)blockIdx.x * blockDim.x + threadIdx.x;
  if (i >= n) return;
  long node = i / HD;
  int col = (int)(i - node * HD);
  long b = node * (3 * HD) + col;
  float ir = gi[b], iz = gi[b + HD], inn = gi[b + 2 * HD];
  float hr = gh[b], hz = gh[b + HD], hn = gh[b + 2 * HD];
  float r = 1.0f / (1.0f + __expf(-(ir + hr)));
  float z = 1.0f / (1.0f + __expf(-(iz + hz)));
  float nv = tanhf(inn + r * hn);
  h[i] = (1.0f - z) * nv + z * h[i];
}

extern "C" void kernel_launch(void* const* d_in, const int* in_sizes, int n_in,
                              void* d_out, int out_size, void* d_ws, size_t ws_size,
                              hipStream_t stream) {
  (void)in_sizes; (void)n_in; (void)out_size; (void)ws_size;
  const float* x     = (const float*)d_in[0];
  const float* msg_W = (const float*)d_in[1];
  const float* msg_b = (const float*)d_in[2];
  const float* w_ih0 = (const float*)d_in[3];
  const float* w_hh0 = (const float*)d_in[4];
  const float* b_ih0 = (const float*)d_in[5];
  const float* b_hh0 = (const float*)d_in[6];
  const float* w_ih1 = (const float*)d_in[7];
  const float* w_hh1 = (const float*)d_in[8];
  const float* b_ih1 = (const float*)d_in[9];
  const float* b_hh1 = (const float*)d_in[10];
  const int*   adj   = (const int*)d_in[11];

  const long NH = (long)ND * HD;
  char* ws = (char*)d_ws;
  size_t off = 0;
  auto alloc = [&](size_t bytes) -> void* {
    void* p = ws + off;
    off = (off + bytes + 255) & ~(size_t)255;
    return p;
  };

  unsigned short* hhi   = (unsigned short*)alloc(NH * 2);
  unsigned short* hlo   = (unsigned short*)alloc(NH * 2);
  unsigned short* xhi   = (unsigned short*)alloc(NH * 2);
  unsigned short* xlo   = (unsigned short*)alloc(NH * 2);
  unsigned short* ichi  = (unsigned short*)alloc(NH * 2);
  unsigned short* iclo  = (unsigned short*)alloc(NH * 2);
  float* incoming = (float*)alloc(NH * 4);
  float* hbuf     = (float*)alloc(NH * 4);
  float* gi       = (float*)alloc(NH * 3 * 4);
  float* gh       = (float*)alloc(NH * 3 * 4);
  const long mWn = 2L * TD * HD * HD;
  unsigned short* mWhi = (unsigned short*)alloc(mWn * 2);
  unsigned short* mWlo = (unsigned short*)alloc(mWn * 2);
  const long wih0n = 3L * HD * HD, whh0n = 3L * HD * HD;
  const long wih1n = 3L * HD * 2 * HD, whh1n = 3L * HD * HD;
  unsigned short* wih0hi = (unsigned short*)alloc(wih0n * 2);
  unsigned short* wih0lo = (unsigned short*)alloc(wih0n * 2);
  unsigned short* whh0hi = (unsigned short*)alloc(whh0n * 2);
  unsigned short* whh0lo = (unsigned short*)alloc(whh0n * 2);
  unsigned short* wih1hi = (unsigned short*)alloc(wih1n * 2);
  unsigned short* wih1lo = (unsigned short*)alloc(wih1n * 2);
  unsigned short* whh1hi = (unsigned short*)alloc(whh1n * 2);
  unsigned short* whh1lo = (unsigned short*)alloc(whh1n * 2);

  auto cvt = [&](const float* s, unsigned short* hi, unsigned short* lo, long n) {
    k_cvt_split<<<dim3((unsigned)((n + 255) / 256)), 256, 0, stream>>>(s, hi, lo, n);
  };

  // one-time weight / input conversions (re-done every call: deterministic, input-only)
  cvt(msg_W, mWhi, mWlo, mWn);
  cvt(w_ih0, wih0hi, wih0lo, wih0n);
  cvt(w_hh0, whh0hi, whh0lo, whh0n);
  cvt(w_ih1, wih1hi, wih1lo, wih1n);
  cvt(w_hh1, whh1hi, whh1lo, whh1n);
  cvt(x, xhi, xlo, NH);
  hipMemcpyAsync(hbuf, x, NH * 4, hipMemcpyDeviceToDevice, stream);

  const dim3 msgGrid((ED + 63) / 64, HD / 64, TD);
  const dim3 gemmGrid((ND + 63) / 64, (3 * HD) / 64);
  const unsigned ewBlocks = (unsigned)((NH + 255) / 256);

  for (int layer = 0; layer < 2; ++layer) {
    const unsigned short* lWhi = mWhi + (size_t)layer * TD * HD * HD;
    const unsigned short* lWlo = mWlo + (size_t)layer * TD * HD * HD;
    const float* lb = msg_b + (size_t)layer * TD * HD;
    for (int step = 0; step < 2; ++step) {
      cvt(hbuf, hhi, hlo, NH);
      k_fill0<<<ewBlocks, 256, 0, stream>>>(incoming, NH);
      k_msg<<<msgGrid, 256, 0, stream>>>(hhi, hlo, lWhi, lWlo, lb, adj, incoming);
      cvt(incoming, ichi, iclo, NH);
      if (layer == 0) {
        k_gemm<<<gemmGrid, 256, 0, stream>>>(ichi, iclo, ichi, iclo, HD, HD,
                                             wih0hi, wih0lo, b_ih0, gi, 3 * HD, ND);
        k_gemm<<<gemmGrid, 256, 0, stream>>>(hhi, hlo, hhi, hlo, HD, HD,
                                             whh0hi, whh0lo, b_hh0, gh, 3 * HD, ND);
      } else {
        k_gemm<<<gemmGrid, 256, 0, stream>>>(xhi, xlo, ichi, iclo, HD, 2 * HD,
                                             wih1hi, wih1lo, b_ih1, gi, 3 * HD, ND);
        k_gemm<<<gemmGrid, 256, 0, stream>>>(hhi, hlo, hhi, hlo, HD, HD,
                                             whh1hi, whh1lo, b_hh1, gh, 3 * HD, ND);
      }
      k_gru<<<ewBlocks, 256, 0, stream>>>(gi, gh, hbuf, NH);
    }
  }
  hipMemcpyAsync(d_out, hbuf, NH * 4, hipMemcpyDeviceToDevice, stream);
}
